// GRUTCMotionForecast_17970143167174
// MI455X (gfx1250) — matrix operand order repaired and checked by reading, not compiled, not run on hardware
//
#include <hip/hip_runtime.h>
#include <hip/hip_bf16.h>

// ---------------------------------------------------------------------------
// GRU + Transformer motion forecast, CDNA5 (gfx1250) WMMA implementation.
// B=2048, T=120, E=144, NH=8, HD=18, FF=576, L=4, H=144, GIN=288.
// - All GEMM inputs stored f16 (halves dominant HBM traffic; WMMA eats f16)
// - fp32 residual stream + fp32 LN/softmax/gates for accuracy
// - guard-free, double-buffered LDS staging; async-to-LDS when available
// ---------------------------------------------------------------------------

typedef _Float16 v16h __attribute__((ext_vector_type(16)));
typedef float    v8f  __attribute__((ext_vector_type(8)));
typedef _Float16 h8   __attribute__((ext_vector_type(8)));
typedef int      v4i  __attribute__((ext_vector_type(4)));

static constexpr int kB   = 2048;
static constexpr int kT   = 120;
static constexpr int kE   = 144;
static constexpr int kEp  = 160;        // padded K / activation row pitch
static constexpr int kNH  = 8;
static constexpr int kHD  = 18;
static constexpr int kFF  = 576;
static constexpr int kL   = 4;
static constexpr int kH   = 144;
static constexpr int kM   = kB * kT;    // 245760 rows
static constexpr int kFUT = 24;
static constexpr int k3E  = 3 * kE;     // 432
static constexpr int k3Ep = 448;        // 432 -> /64
static constexpr int kEp64  = 192;      // 144 -> /64
static constexpr int kFFp64 = 576;      // already /64

// ---------------------------------------------------------------------------
// Async global->LDS copy (CDNA5 GLOBAL_LOAD_ASYNC_TO_LDS_B128, ASYNCcnt),
// with synchronous fallback when the builtin is unavailable.
// Probe result: builtin exists with 4 params; param0 = v4i in AS1 (global),
// param1 = v4i in AS3 (LDS), then imm offset + imm cpol.
// ---------------------------------------------------------------------------
#if defined(__has_builtin)
#if __has_builtin(__builtin_amdgcn_global_load_async_to_lds_b128)
#define HAVE_ASYNC_LDS 1
#endif
#endif
#ifndef HAVE_ASYNC_LDS
#define HAVE_ASYNC_LDS 0
#endif

__device__ __forceinline__ void cp16_async(const _Float16* g, _Float16* l)
{
#if HAVE_ASYNC_LDS
    __builtin_amdgcn_global_load_async_to_lds_b128(
        (__attribute__((address_space(1))) v4i*)(v4i*)g,
        (__attribute__((address_space(3))) v4i*)(v4i*)l, 0, 0);
#else
    *(h8*)l = *(const h8*)g;
#endif
}

__device__ __forceinline__ void async_fence()
{
#if HAVE_ASYNC_LDS
    asm volatile("s_wait_asynccnt 0x0" ::: "memory");
#endif
}

// ---------------------------------------------------------------------------
// f32 -> f16 conversion with [N,K] -> [Np,Kp] zero padding (weights, one-shot)
// ---------------------------------------------------------------------------
__global__ void cvt_pad(const float* __restrict__ src, int srcN, int srcK,
                        int src_ld, int src_off,
                        _Float16* __restrict__ dst, int Np, int Kp)
{
    int i = blockIdx.x * blockDim.x + threadIdx.x;
    if (i >= Np * Kp) return;
    int n = i / Kp, k = i % Kp;
    dst[i] = (n < srcN && k < srcK)
                 ? (_Float16)src[(size_t)n * src_ld + src_off + k]
                 : (_Float16)0.f;
}

// ---------------------------------------------------------------------------
// seed [M,144] -> xbuf f32 [M,160] + xh f16 [M,160], zero pad cols
// ---------------------------------------------------------------------------
__global__ void seed_to_x(const float* __restrict__ seed,
                          float* __restrict__ x, _Float16* __restrict__ xh)
{
    int i = blockIdx.x * blockDim.x + threadIdx.x;
    if (i >= kM * kEp) return;
    int row = i / kEp, e = i % kEp;
    float v = (e < kE) ? seed[(size_t)row * kE + e] : 0.f;
    x[i]  = v;
    xh[i] = (_Float16)v;
}

// ---------------------------------------------------------------------------
// WMMA GEMM:  C[M,N] = act( A[M,Kp](f16) * W[Np,Kp](f16)^T + bias )
// Kp % 32 == 0, M % 64 == 0, W zero-padded to /64 rows.
// 256 threads (8 waves), 64x64 tile, double-buffered LDS, async staging.
// ---------------------------------------------------------------------------
template <typename OutT>
__global__ __launch_bounds__(256)
void wmma_gemm(const _Float16* __restrict__ A, int lda,
               const _Float16* __restrict__ W, int ldw,
               const float* __restrict__ bias,
               OutT* __restrict__ C, int ldc,
               int N, int Kp, int relu)
{
    __shared__ _Float16 As[2][64][40];   // pitch 40: 16B-aligned, bank-spread
    __shared__ _Float16 Bs[2][64][40];   // Bs[n][k] == W[n0+n][kk+k]

    const int tid  = threadIdx.x;
    const int lane = tid & 31;
    const int wave = tid >> 5;
    const int wr = wave & 3;
    const int wc = wave >> 2;
    const int lr = lane & 15;
    const int lh = lane >> 4;
    const int m0 = blockIdx.y * 64;
    const int n0 = blockIdx.x * 64;
    const int srow = tid >> 2;
    const int sseg = (tid & 3) * 8;

    const _Float16* Ald = A + (size_t)(m0 + srow) * lda + sseg;
    const _Float16* Wld = W + (size_t)(n0 + srow) * ldw + sseg;

    v8f acc0 = {}; v8f acc1 = {};

    // prologue: stage K-tile 0
    cp16_async(Ald, &As[0][srow][sseg]);
    cp16_async(Wld, &Bs[0][srow][sseg]);
    async_fence();
    __syncthreads();

    int buf = 0;
    for (int kk = 32; kk <= Kp; kk += 32) {
        const bool more = (kk < Kp);

#if HAVE_ASYNC_LDS
        if (more) {                       // async copy next tile into other bank
            cp16_async(Ald + kk, &As[buf ^ 1][srow][sseg]);
            cp16_async(Wld + kk, &Bs[buf ^ 1][srow][sseg]);
            if (kk + 32 < Kp) {
                __builtin_prefetch(Ald + kk + 32, 0, 1);
                __builtin_prefetch(Wld + kk + 32, 0, 1);
            }
        }
#else
        h8 na, nb;
        if (more) {                       // issue next tile's loads early
            na = *(const h8*)(Ald + kk);
            nb = *(const h8*)(Wld + kk);
            if (kk + 32 < Kp) {
                __builtin_prefetch(Ald + kk + 32, 0, 1);
                __builtin_prefetch(Wld + kk + 32, 0, 1);
            }
        }
#endif

        // fragments from current buffer (same contiguous k = 16*lh + i order in A and B)
        v16h a;
        const _Float16* ap = &As[buf][wr * 16 + lr][lh * 16];
        #pragma unroll
        for (int i = 0; i < 16; ++i) a[i] = ap[i];

        v16h b0, b1;
        const _Float16* bp0 = &Bs[buf][wc * 32 + lr][lh * 16];
        const _Float16* bp1 = &Bs[buf][wc * 32 + 16 + lr][lh * 16];
        #pragma unroll
        for (int j = 0; j < 16; ++j) { b0[j] = bp0[j]; b1[j] = bp1[j]; }

        acc0 = __builtin_amdgcn_wmma_f32_16x16x32_f16(false, a, false, b0, (short)0, acc0, false, false);
        acc1 = __builtin_amdgcn_wmma_f32_16x16x32_f16(false, a, false, b1, (short)0, acc1, false, false);

#if HAVE_ASYNC_LDS
        async_fence();                    // this wave's async copies landed
#else
        if (more) {
            *(h8*)&As[buf ^ 1][srow][sseg] = na;
            *(h8*)&Bs[buf ^ 1][srow][sseg] = nb;
        }
#endif
        __syncthreads();
        buf ^= 1;
    }

    // epilogue (only place with N guards)
    const int ng0 = n0 + wc * 32 + lr;
    const int ng1 = ng0 + 16;
    float bv0 = 0.f, bv1 = 0.f;
    if (bias) {
        if (ng0 < N) bv0 = bias[ng0];
        if (ng1 < N) bv1 = bias[ng1];
    }
    #pragma unroll
    for (int r = 0; r < 8; ++r) {
        const int m = m0 + wr * 16 + r + lh * 8;
        float v0 = acc0[r] + bv0;
        float v1 = acc1[r] + bv1;
        if (relu) { v0 = fmaxf(v0, 0.f); v1 = fmaxf(v1, 0.f); }
        if (ng0 < N) C[(size_t)m * ldc + ng0] = (OutT)v0;
        if (ng1 < N) C[(size_t)m * ldc + ng1] = (OutT)v1;
    }
}

// ---------------------------------------------------------------------------
// Fused attention: one block per (batch, head). qkv f16 in, o f16 out.
// T=120 padded to 128, HD=18 padded to 32; softmax fp32 in LDS.
// ---------------------------------------------------------------------------
__global__ __launch_bounds__(256)
void attention(const _Float16* __restrict__ qkv, _Float16* __restrict__ o)
{
    __shared__ _Float16 Qs[128][40];
    __shared__ _Float16 Ks[128][40];
    __shared__ _Float16 Vt[32][136];   // V transposed: [d][t]
    __shared__ float    Sm[128][132];  // scores -> probabilities in place

    const int bh = blockIdx.x;
    const int b = bh >> 3, h = bh & 7;
    const int tid  = threadIdx.x;
    const int lane = tid & 31;
    const int wave = tid >> 5;
    const int lr = lane & 15;
    const int lh = lane >> 4;
    const size_t base = (size_t)b * kT * k3E;
    const int qo = h * kHD, ko = kE + h * kHD, vo = 2 * kE + h * kHD;

    // staging: 2 threads per row; d<18 predicate is compile-time per lane
    {
        const int t = tid >> 1;
        const int d0 = (tid & 1) * 16;
        if (t < kT) {
            const _Float16* row = qkv + base + (size_t)t * k3E;
            #pragma unroll
            for (int i = 0; i < 16; ++i) {
                const int d = d0 + i;
                _Float16 q = (d < kHD) ? row[qo + d] : (_Float16)0.f;
                _Float16 k = (d < kHD) ? row[ko + d] : (_Float16)0.f;
                _Float16 v = (d < kHD) ? row[vo + d] : (_Float16)0.f;
                Qs[t][d] = q; Ks[t][d] = k; Vt[d][t] = v;
            }
        } else {
            #pragma unroll
            for (int i = 0; i < 16; ++i) {
                const int d = d0 + i;
                Qs[t][d] = (_Float16)0.f;
                Ks[t][d] = (_Float16)0.f;
                Vt[d][t] = (_Float16)0.f;
            }
        }
    }
    __syncthreads();

    // S = (Q K^T) / sqrt(HD); each wave owns 16 query rows
    {
        v16h a;
        const _Float16* ap = &Qs[wave * 16 + lr][lh * 16];
        #pragma unroll
        for (int i = 0; i < 16; ++i) a[i] = ap[i];
        const float scale = 0.2357022603955158f;  // 1/sqrt(18)
        #pragma unroll
        for (int nt = 0; nt < 8; ++nt) {
            v16h bf;
            const _Float16* bp = &Ks[nt * 16 + lr][lh * 16];
            #pragma unroll
            for (int j = 0; j < 16; ++j) bf[j] = bp[j];
            v8f acc = {};
            acc = __builtin_amdgcn_wmma_f32_16x16x32_f16(false, a, false, bf, (short)0, acc, false, false);
            const int n = nt * 16 + lr;
            #pragma unroll
            for (int r = 0; r < 8; ++r)
                Sm[wave * 16 + r + lh * 8][n] = acc[r] * scale;
        }
    }
    __syncthreads();

    // softmax over valid keys (rows >=120 hold zeros -> harmless)
    if (tid < kT) {
        float mx = -1e30f;
        for (int n = 0; n < kT; ++n) mx = fmaxf(mx, Sm[tid][n]);
        float s = 0.f;
        for (int n = 0; n < kT; ++n) { float e = __expf(Sm[tid][n] - mx); Sm[tid][n] = e; s += e; }
        const float inv = 1.f / s;
        for (int n = 0; n < kT; ++n) Sm[tid][n] *= inv;
        for (int n = kT; n < 128; ++n) Sm[tid][n] = 0.f;  // padded K cols must be 0
    }
    __syncthreads();

    // O = P V
    {
        v8f acc0 = {}, acc1 = {};
        #pragma unroll
        for (int kk = 0; kk < 128; kk += 32) {
            v16h a;
            const float* sp = &Sm[wave * 16 + lr][kk + lh * 16];
            #pragma unroll
            for (int i = 0; i < 16; ++i) a[i] = (_Float16)sp[i];
            v16h b0, b1;
            const _Float16* vp0 = &Vt[lr][kk + lh * 16];
            const _Float16* vp1 = &Vt[16 + lr][kk + lh * 16];
            #pragma unroll
            for (int j = 0; j < 16; ++j) { b0[j] = vp0[j]; b1[j] = vp1[j]; }
            acc0 = __builtin_amdgcn_wmma_f32_16x16x32_f16(false, a, false, b0, (short)0, acc0, false, false);
            acc1 = __builtin_amdgcn_wmma_f32_16x16x32_f16(false, a, false, b1, (short)0, acc1, false, false);
        }
        const int d0 = lr, d1 = 16 + lr;
        #pragma unroll
        for (int r = 0; r < 8; ++r) {
            const int m = wave * 16 + r + lh * 8;
            if (m < kT) {
                _Float16* orow = o + ((size_t)b * kT + m) * kEp + h * kHD;
                orow[d0] = (_Float16)acc0[r];
                if (d1 < kHD) orow[d1] = (_Float16)acc1[r];
            }
        }
    }
}

// ---------------------------------------------------------------------------
// x = LayerNorm(x + y) in fp32; writes fp32 master and f16 mirror
// x/xh pitch kEp, y pitch kE
// ---------------------------------------------------------------------------
__global__ void add_ln(float* __restrict__ x, _Float16* __restrict__ xh,
                       const float* __restrict__ y,
                       const float* __restrict__ w, const float* __restrict__ bb, int M)
{
    const int row = blockIdx.x * blockDim.x + threadIdx.x;
    if (row >= M) return;
    float* xr = x + (size_t)row * kEp;
    _Float16* xhr = xh + (size_t)row * kEp;
    const float* yr = y + (size_t)row * kE;
    float s = 0.f, ss = 0.f;
    for (int e = 0; e < kE; ++e) { float v = xr[e] + yr[e]; s += v; ss += v * v; }
    const float m   = s * (1.f / kE);
    const float var = ss * (1.f / kE) - m * m;
    const float inv = rsqrtf(var + 1e-5f);
    for (int e = 0; e < kE; ++e) {
        float v = (xr[e] + yr[e] - m) * inv * w[e] + bb[e];
        xr[e]  = v;
        xhr[e] = (_Float16)v;
    }
}

// ---------------------------------------------------------------------------
// xmh[b][e] = mean_t x[b][t][e]  (f32 in pitch kEp, f16 out pitch kEp)
// ---------------------------------------------------------------------------
__global__ void mean_t(const float* __restrict__ x, _Float16* __restrict__ xmh)
{
    const int idx = blockIdx.x * blockDim.x + threadIdx.x;
    if (idx >= kB * kE) return;
    const int b = idx / kE, e = idx % kE;
    const float* p = x + (size_t)b * kT * kEp + e;
    float s = 0.f;
    for (int t = 0; t < kT; ++t) s += p[(size_t)t * kEp];
    xmh[(size_t)b * kEp + e] = (_Float16)(s * (1.f / kT));
}

// ---------------------------------------------------------------------------
// GRU init: inp = seed[:, -1] as f16 (pad cols pre-zeroed)
// ---------------------------------------------------------------------------
__global__ void gru_init(const float* __restrict__ seed, _Float16* __restrict__ inph)
{
    const int idx = blockIdx.x * blockDim.x + threadIdx.x;
    if (idx >= kB * kE) return;
    const int b = idx / kE, e = idx % kE;
    inph[(size_t)b * kEp + e] = (_Float16)seed[((size_t)b * kT + (kT - 1)) * kE + e];
}

// ---------------------------------------------------------------------------
// GRU gates (fp32): h = (1-z)*n + z*h ; writes fp32 h and f16 mirror
// ---------------------------------------------------------------------------
__global__ void gru_gate(const float* __restrict__ gi, const float* __restrict__ gctx,
                         const float* __restrict__ gh,
                         float* __restrict__ hf, _Float16* __restrict__ hh)
{
    const int idx = blockIdx.x * blockDim.x + threadIdx.x;
    if (idx >= kB * kH) return;
    const int b = idx / kH, j = idx % kH;
    const size_t o = (size_t)b * k3E;
    const float ir = gi[o + j]          + gctx[o + j];
    const float iz = gi[o + kH + j]     + gctx[o + kH + j];
    const float in = gi[o + 2 * kH + j] + gctx[o + 2 * kH + j];
    const float hr = gh[o + j];
    const float hz = gh[o + kH + j];
    const float hn = gh[o + 2 * kH + j];
    const float r = 1.f / (1.f + __expf(-(ir + hr)));
    const float z = 1.f / (1.f + __expf(-(iz + hz)));
    const float n = tanhf(in + r * hn);
    const float hold = hf[(size_t)b * kH + j];
    const float hnew = (1.f - z) * n + z * hold;
    hf[(size_t)b * kH + j]  = hnew;
    hh[(size_t)b * kEp + j] = (_Float16)hnew;
}

// ---------------------------------------------------------------------------
// rot f32 [2048,144] -> d_out[b,step,:] (fp32) and next inp (f16)
// ---------------------------------------------------------------------------
__global__ void emit_rot(const float* __restrict__ rot, float* __restrict__ out,
                         _Float16* __restrict__ inph, int step)
{
    const int idx = blockIdx.x * blockDim.x + threadIdx.x;
    if (idx >= kB * kE) return;
    const int b = idx / kE, e = idx % kE;
    const float v = rot[(size_t)b * kE + e];
    out[((size_t)b * kFUT + step) * kE + e] = v;
    inph[(size_t)b * kEp + e] = (_Float16)v;
}

// ---------------------------------------------------------------------------
// host-side orchestration
// ---------------------------------------------------------------------------
extern "C" void kernel_launch(void* const* d_in, const int* in_sizes, int n_in,
                              void* d_out, int out_size, void* d_ws, size_t ws_size,
                              hipStream_t stream)
{
    const float* seed   = (const float*)d_in[0];
    const float* qkv_w  = (const float*)d_in[1];
    const float* qkv_b  = (const float*)d_in[2];
    const float* out_w  = (const float*)d_in[3];
    const float* out_b  = (const float*)d_in[4];
    const float* ln1_w  = (const float*)d_in[5];
    const float* ln1_b  = (const float*)d_in[6];
    const float* f1_w   = (const float*)d_in[7];
    const float* f1_b   = (const float*)d_in[8];
    const float* f2_w   = (const float*)d_in[9];
    const float* f2_b   = (const float*)d_in[10];
    const float* ln2_w  = (const float*)d_in[11];
    const float* ln2_b  = (const float*)d_in[12];
    const float* proj_w = (const float*)d_in[13];
    const float* proj_b = (const float*)d_in[14];
    const float* w_ih   = (const float*)d_in[15];
    const float* w_hh   = (const float*)d_in[16];
    const float* b_ih   = (const float*)d_in[17];
    const float* b_hh   = (const float*)d_in[18];
    const float* spl_w  = (const float*)d_in[19];
    const float* spl_b  = (const float*)d_in[20];

    char* ws = (char*)d_ws;
    size_t off = 0;
    auto carve = [&](size_t bytes) -> char* {
        char* p = ws + off;
        off += (bytes + 255) & ~(size_t)255;
        return p;
    };

    // padded f16 weights [Np, Kp]
    _Float16* qkvh = (_Float16*)carve((size_t)kL * k3Ep * kEp * 2);
    _Float16* outh = (_Float16*)carve((size_t)kL * kEp64 * kEp * 2);
    _Float16* f1h  = (_Float16*)carve((size_t)kL * kFFp64 * kEp * 2);
    _Float16* f2h  = (_Float16*)carve((size_t)kL * kEp64 * kFF * 2);
    _Float16* prh  = (_Float16*)carve((size_t)kEp64 * kEp * 2);
    _Float16* wihc = (_Float16*)carve((size_t)k3Ep * kEp * 2);   // w_ih[:,144:288]
    _Float16* wihx = (_Float16*)carve((size_t)k3Ep * kEp * 2);   // w_ih[:,0:144]
    _Float16* whh  = (_Float16*)carve((size_t)k3Ep * kEp * 2);
    _Float16* splh = (_Float16*)carve((size_t)kEp64 * kEp * 2);

    // activations
    float*    xbuf  = (float*)   carve((size_t)kM * kEp * 4);   // fp32 residual
    _Float16* xh    = (_Float16*)carve((size_t)kM * kEp * 2);   // f16 mirror
    _Float16* bigh  = (_Float16*)carve((size_t)kM * kFF * 2);   // qkv then ff
    _Float16* obufh = (_Float16*)carve((size_t)kM * kEp * 2);   // attention out
    float*    ybuf  = (float*)   carve((size_t)kM * kE  * 4);   // GEMM fp32 out
    _Float16* xmh   = (_Float16*)carve((size_t)kB * kEp * 2);
    _Float16* ctxh  = (_Float16*)carve((size_t)kB * kEp * 2);
    float*    gctx  = (float*)   carve((size_t)kB * k3E * 4);
    float*    gib   = (float*)   carve((size_t)kB * k3E * 4);
    float*    ghb   = (float*)   carve((size_t)kB * k3E * 4);
    float*    hf    = (float*)   carve((size_t)kB * kH  * 4);
    _Float16* hh    = (_Float16*)carve((size_t)kB * kEp * 2);
    _Float16* inph  = (_Float16*)carve((size_t)kB * kEp * 2);
    float*    rotb  = (float*)   carve((size_t)kB * kE  * 4);

    // zero pad columns / initial GRU state
    (void)hipMemsetAsync(obufh, 0, (size_t)kM * kEp * 2, stream);
    (void)hipMemsetAsync(xmh,   0, (size_t)kB * kEp * 2, stream);
    (void)hipMemsetAsync(ctxh,  0, (size_t)kB * kEp * 2, stream);
    (void)hipMemsetAsync(hf,    0, (size_t)kB * kH  * 4, stream);
    (void)hipMemsetAsync(hh,    0, (size_t)kB * kEp * 2, stream);
    (void)hipMemsetAsync(inph,  0, (size_t)kB * kEp * 2, stream);

    auto cvt = [&](const float* s, int srcN, int srcK, int sld, int soff,
                   _Float16* d, int Np, int Kp) {
        int n = Np * Kp;
        cvt_pad<<<(n + 255) / 256, 256, 0, stream>>>(s, srcN, srcK, sld, soff, d, Np, Kp);
    };
    for (int l = 0; l < kL; ++l) {
        cvt(qkv_w + (size_t)l * k3E * kE, k3E, kE, kE, 0, qkvh + (size_t)l * k3Ep * kEp, k3Ep, kEp);
        cvt(out_w + (size_t)l * kE * kE,  kE,  kE, kE, 0, outh + (size_t)l * kEp64 * kEp, kEp64, kEp);
        cvt(f1_w  + (size_t)l * kFF * kE, kFF, kE, kE, 0, f1h  + (size_t)l * kFFp64 * kEp, kFFp64, kEp);
        cvt(f2_w  + (size_t)l * kE * kFF, kE, kFF, kFF, 0, f2h + (size_t)l * kEp64 * kFF, kEp64, kFF);
    }
    cvt(proj_w, kE, kE, kE, 0, prh, kEp64, kEp);
    cvt(w_ih, k3E, kE, 2 * kE, kE, wihc, k3Ep, kEp);
    cvt(w_ih, k3E, kE, 2 * kE, 0,  wihx, k3Ep, kEp);
    cvt(w_hh, k3E, kH, kH, 0, whh, k3Ep, kEp);
    cvt(spl_w, kE, kH, kH, 0, splh, kEp64, kEp);

    auto gemm_h = [&](const _Float16* A, int lda, const _Float16* W, int ldw,
                      const float* bias, _Float16* C, int ldc,
                      int M, int N, int Kp, int relu) {
        dim3 grid((N + 63) / 64, M / 64);
        wmma_gemm<_Float16><<<grid, 256, 0, stream>>>(A, lda, W, ldw, bias, C, ldc, N, Kp, relu);
    };
    auto gemm_f = [&](const _Float16* A, int lda, const _Float16* W, int ldw,
                      const float* bias, float* C, int ldc,
                      int M, int N, int Kp, int relu) {
        dim3 grid((N + 63) / 64, M / 64);
        wmma_gemm<float><<<grid, 256, 0, stream>>>(A, lda, W, ldw, bias, C, ldc, N, Kp, relu);
    };

    // x0 = seed
    seed_to_x<<<(kM * kEp + 255) / 256, 256, 0, stream>>>(seed, xbuf, xh);

    for (int l = 0; l < kL; ++l) {
        const _Float16* qwl = qkvh + (size_t)l * k3Ep * kEp;
        const _Float16* owl = outh + (size_t)l * kEp64 * kEp;
        const _Float16* f1l = f1h  + (size_t)l * kFFp64 * kEp;
        const _Float16* f2l = f2h  + (size_t)l * kEp64 * kFF;

        // QKV (f16 out) -> bigh[:, 0:432]
        gemm_h(xh, kEp, qwl, kEp, qkv_b + (size_t)l * k3E, bigh, k3E, kM, k3E, kEp, 0);
        // attention -> obufh (f16, pad cols stay zero)
        attention<<<kB * kNH, 256, 0, stream>>>(bigh, obufh);
        // out-proj (f32 out) -> ybuf
        gemm_f(obufh, kEp, owl, kEp, out_b + (size_t)l * kE, ybuf, kE, kM, kE, kEp, 0);
        // x = LN(x + y): fp32 master + f16 mirror
        add_ln<<<(kM + 255) / 256, 256, 0, stream>>>(xbuf, xh, ybuf,
            ln1_w + (size_t)l * kE, ln1_b + (size_t)l * kE, kM);
        // FFN1 (relu, f16 out) -> bigh[:, 0:576]  (qkv dead)
        gemm_h(xh, kEp, f1l, kEp, f1_b + (size_t)l * kFF, bigh, kFF, kM, kFF, kEp, 1);
        // FFN2 (f32 out) -> ybuf
        gemm_f(bigh, kFF, f2l, kFF, f2_b + (size_t)l * kE, ybuf, kE, kM, kE, kFF, 0);
        // x = LN(x + ff)
        add_ln<<<(kM + 255) / 256, 256, 0, stream>>>(xbuf, xh, ybuf,
            ln2_w + (size_t)l * kE, ln2_b + (size_t)l * kE, kM);
    }

    // ctx = mean_t(x) @ proj_w^T + proj_b  (f16 out for next GEMM)
    mean_t<<<(kB * kE + 255) / 256, 256, 0, stream>>>(xbuf, xmh);
    gemm_h(xmh, kEp, prh, kEp, proj_b, ctxh, kEp, kB, kE, kEp, 0);

    // gctx = ctx @ w_ih[:,144:288]^T + b_ih  (step-invariant gate part, fp32)
    gemm_f(ctxh, kEp, wihc, kEp, b_ih, gctx, k3E, kB, k3E, kEp, 0);

    // GRU decode: batched WMMA GEMMs + fp32 gate fusion per step
    gru_init<<<(kB * kE + 255) / 256, 256, 0, stream>>>(seed, inph);
    float* out = (float*)d_out;
    for (int s = 0; s < kFUT; ++s) {
        gemm_f(inph, kEp, wihx, kEp, nullptr, gib, k3E, kB, k3E, kEp, 0);
        gemm_f(hh,   kEp, whh,  kEp, b_hh,    ghb, k3E, kB, k3E, kEp, 0);
        gru_gate<<<(kB * kH + 255) / 256, 256, 0, stream>>>(gib, gctx, ghb, hf, hh);
        gemm_f(hh, kEp, splh, kEp, spl_b, rotb, kE, kB, kE, kEp, 0);
        emit_rot<<<(kB * kE + 255) / 256, 256, 0, stream>>>(rotb, out, inph, s);
    }
}
